// AMNet_36490042146906
// MI455X (gfx1250) — matrix operand order repaired and checked by reading, not compiled, not run on hardware
//
#include <hip/hip_runtime.h>
#include <hip/hip_bf16.h>

typedef __attribute__((ext_vector_type(16))) __bf16 v16bf;
typedef __attribute__((ext_vector_type(8)))  __bf16 v8bf;
typedef __attribute__((ext_vector_type(8)))  float  v8f;

#define HID 64
#define INF 128

// W [K,N] fp32 row-major -> Wt [N,K] bf16 (so B fragments are contiguous in K)
__global__ void cvt_transpose(const float* __restrict__ w, __bf16* __restrict__ wt, int K, int N) {
    int i = blockIdx.x * blockDim.x + threadIdx.x;
    if (i < K * N) {
        int k = i / N, n = i % N;
        wt[(size_t)n * K + k] = (__bf16)w[i];
    }
}

__global__ void zero_f32(float* __restrict__ p, int n) {
    int i = blockIdx.x * blockDim.x + threadIdx.x;
    for (; i < n; i += gridDim.x * blockDim.x) p[i] = 0.0f;
}

// ---------------- fused MLP: h = relu(x@W1+b1)@W2 + b2 ----------------
// Block = 256 threads = 8 waves; block covers 32 output rows.
// Wave w -> (rowTile = w>>2, colTile = w&3): one 16x16 WMMA tile each.
// GEMM1 (K=128): A = x fp32 converted inline to bf16; result (relu) staged in LDS.
// GEMM2 (K=64):  A fragments read back from LDS (ds_load_b128); output fp32.
__global__ __launch_bounds__(256) void mlp_fused(const float* __restrict__ x,
                                                 const __bf16* __restrict__ w1t,
                                                 const float* __restrict__ b1,
                                                 const __bf16* __restrict__ w2t,
                                                 const float* __restrict__ b2,
                                                 float* __restrict__ h,
                                                 int M) {
    __shared__ __bf16 h1s[32 * HID];   // 4 KB: relu(x@W1+b1) tile, bf16

    const int lane  = threadIdx.x & 31;
    const int wave  = threadIdx.x >> 5;
    const int half  = lane >> 4;        // 0: lanes 0-15, 1: lanes 16-31
    const int l16   = lane & 15;
    const int mTile = wave >> 2;        // 0..1
    const int nBase = (wave & 3) * 16;  // 0,16,32,48
    const int mBase = blockIdx.x * 32 + mTile * 16;
    if (mBase + 16 > M) return;

    // ---- GEMM1: 16x16 tile of relu(x@W1+b1), K=128, inline fp32->bf16 ----
    {
        v8f c = {};
        const float*  arow = x   + (size_t)(mBase + l16) * INF;
        const __bf16* bcol = w1t + (size_t)(nBase + l16) * INF;
#pragma unroll
        for (int kk = 0; kk < INF; kk += 32) {
            // A 16x32 bf16, k order chosen here (same as B): half=0 holds K kk+0..15; half=1 kk+16..31
            v8f a0 = *(const v8f*)(arow + kk + half * 16);
            v8f a1 = *(const v8f*)(arow + kk + half * 16 + 8);
            union { v16bf v; v8bf g[2]; } a;
#pragma unroll
            for (int i = 0; i < 8; ++i) {
                a.g[0][i] = (__bf16)a0[i];
                a.g[1][i] = (__bf16)a1[i];
            }
            // B 32x16, same k order as A: lane(col) half=0 holds K kk+0..15; half=1 kk+16..31
            v16bf b = *(const v16bf*)(bcol + kk + half * 16);
            c = __builtin_amdgcn_wmma_f32_16x16x32_bf16(false, a.v, false, b,
                                                        (short)0, c, false, false);
        }
        // C layout: VGPR j -> row (half*8 + j), col l16 ; stage relu result in LDS
        const float bv = b1[nBase + l16];
#pragma unroll
        for (int j = 0; j < 8; ++j) {
            float v = fmaxf(c[j] + bv, 0.0f);
            h1s[(mTile * 16 + half * 8 + j) * HID + nBase + l16] = (__bf16)v;
        }
    }

    __syncthreads();

    // ---- GEMM2: 16x16 tile of h1 @ W2 + b2, K=64, A from LDS ----
    {
        v8f c = {};
        const __bf16* arow = h1s + (mTile * 16 + l16) * HID;
        const __bf16* bcol = w2t + (size_t)(nBase + l16) * HID;
#pragma unroll
        for (int kk = 0; kk < HID; kk += 32) {
            union { v16bf v; v8bf g[2]; } a;
            a.g[0] = *(const v8bf*)(arow + kk + half * 16);       // ds_load_b128
            a.g[1] = *(const v8bf*)(arow + kk + half * 16 + 8);
            v16bf b = *(const v16bf*)(bcol + kk + half * 16);
            c = __builtin_amdgcn_wmma_f32_16x16x32_bf16(false, a.v, false, b,
                                                        (short)0, c, false, false);
        }
        const float bv = b2[nBase + l16];
#pragma unroll
        for (int j = 0; j < 8; ++j) {
            size_t o = (size_t)(mBase + half * 8 + j) * HID + nBase + l16;
            h[o] = c[j] + bv;
        }
    }
}

// ---------------- per-node inverse L2 norm (wave per node) ----------------
__global__ __launch_bounds__(256) void node_rnorm(const float* __restrict__ h,
                                                  float* __restrict__ rnorm, int Nn) {
    int node = blockIdx.x * 8 + (threadIdx.x >> 5);
    if (node >= Nn) return;
    int lane = threadIdx.x & 31;
    float v0 = h[(size_t)node * HID + lane];
    float v1 = h[(size_t)node * HID + 32 + lane];
    float ss = v0 * v0 + v1 * v1;
#pragma unroll
    for (int off = 16; off > 0; off >>= 1) ss += __shfl_xor(ss, off, 32);
    if (lane == 0) rnorm[node] = 1.0f / fmaxf(sqrtf(ss), 1e-12f);
}

// ---------------- edge pass 1: ex[e] = exp(beta * cos(h_dst, h_src)); s[dst] += ex ----------------
// cos in [-1,1] and softmax is shift-invariant -> no segment-max pass needed.
__global__ __launch_bounds__(256) void edge_exp(const float* __restrict__ h,
                                                const float* __restrict__ rnorm,
                                                const int* __restrict__ eidx,
                                                int E, int Etot,
                                                const float* __restrict__ beta,
                                                float* __restrict__ exbuf,
                                                float* __restrict__ ssum) {
    int e = blockIdx.x * 8 + (threadIdx.x >> 5);
    if (e >= Etot) return;
    int lane = threadIdx.x & 31;
    int s, d;
    if (e < E) { s = eidx[e]; d = eidx[E + e]; }
    else       { s = d = e - E; }   // self loop
    const float* hs = h + (size_t)s * HID;
    const float* hd = h + (size_t)d * HID;
    float acc = hd[lane] * hs[lane] + hd[32 + lane] * hs[32 + lane];
#pragma unroll
    for (int off = 16; off > 0; off >>= 1) acc += __shfl_xor(acc, off, 32);
    if (lane == 0) {
        float ex = __expf(beta[0] * acc * rnorm[s] * rnorm[d]);
        exbuf[e] = ex;
        atomicAdd(&ssum[d], ex);
    }
}

// ---------------- edge pass 2: hout[dst] += (ex/s[dst]) * h[src] ----------------
__global__ __launch_bounds__(256) void edge_agg(const float* __restrict__ h,
                                                const int* __restrict__ eidx,
                                                int E, int Etot,
                                                const float* __restrict__ exbuf,
                                                const float* __restrict__ ssum,
                                                float* __restrict__ hout) {
    int e = blockIdx.x * 8 + (threadIdx.x >> 5);
    if (e >= Etot) return;
    int lane = threadIdx.x & 31;
    int s, d;
    if (e < E) { s = eidx[e]; d = eidx[E + e]; }
    else       { s = d = e - E; }
    float alpha = exbuf[e] / (ssum[d] + 1e-16f);
    const float* hs = h + (size_t)s * HID;
    float* hd = hout + (size_t)d * HID;
    atomicAdd(&hd[lane],      alpha * hs[lane]);
    atomicAdd(&hd[32 + lane], alpha * hs[32 + lane]);
}

// ---------------- classifier: y[N,2] = hout @ Wc + bc (wave per node) ----------------
__global__ __launch_bounds__(256) void classifier(const float* __restrict__ hout,
                                                  const float* __restrict__ Wc,
                                                  const float* __restrict__ bc,
                                                  float* __restrict__ y, int Nn) {
    int node = blockIdx.x * 8 + (threadIdx.x >> 5);
    if (node >= Nn) return;
    int lane = threadIdx.x & 31;
    float v0 = hout[(size_t)node * HID + lane];
    float v1 = hout[(size_t)node * HID + 32 + lane];
    float y0 = v0 * Wc[lane * 2 + 0] + v1 * Wc[(lane + 32) * 2 + 0];
    float y1 = v0 * Wc[lane * 2 + 1] + v1 * Wc[(lane + 32) * 2 + 1];
#pragma unroll
    for (int off = 16; off > 0; off >>= 1) {
        y0 += __shfl_xor(y0, off, 32);
        y1 += __shfl_xor(y1, off, 32);
    }
    if (lane == 0) {
        y[(size_t)node * 2 + 0] = y0 + bc[0];
        y[(size_t)node * 2 + 1] = y1 + bc[1];
    }
}

extern "C" void kernel_launch(void* const* d_in, const int* in_sizes, int n_in,
                              void* d_out, int out_size, void* d_ws, size_t ws_size,
                              hipStream_t stream) {
    const float* x    = (const float*)d_in[0];
    const int*   eidx = (const int*)  d_in[1];
    const float* W1   = (const float*)d_in[2];
    const float* b1   = (const float*)d_in[3];
    const float* W2   = (const float*)d_in[4];
    const float* b2   = (const float*)d_in[5];
    const float* beta = (const float*)d_in[6];
    const float* Wc   = (const float*)d_in[7];
    const float* bc   = (const float*)d_in[8];
    float* y = (float*)d_out;

    const int Nn   = in_sizes[0] / INF;   // 100000
    const int E    = in_sizes[1] / 2;     // 1600000
    const int Etot = E + Nn;              // + self loops

    // workspace carve-out (256B aligned)
    char* ws = (char*)d_ws;
    size_t cur = 0;
    auto carve = [&](size_t bytes) -> void* {
        void* p = ws + cur;
        cur = (cur + bytes + 255) & ~(size_t)255;
        return p;
    };
    __bf16* w1t   = (__bf16*)carve((size_t)INF * HID * sizeof(__bf16));
    __bf16* w2t   = (__bf16*)carve((size_t)HID * HID * sizeof(__bf16));
    float*  h     = (float*) carve((size_t)Nn * HID * sizeof(float));
    float*  rnorm = (float*) carve((size_t)Nn * sizeof(float));
    float*  ssum  = (float*) carve((size_t)Nn * sizeof(float));
    float*  exbuf = (float*) carve((size_t)Etot * sizeof(float));
    float*  hout  = (float*) carve((size_t)Nn * HID * sizeof(float));

    // 1) weight transposition + bf16 conversion (tiny)
    cvt_transpose<<<(INF * HID + 255) / 256, 256, 0, stream>>>(W1, w1t, INF, HID);
    cvt_transpose<<<(HID * HID + 255) / 256, 256, 0, stream>>>(W2, w2t, HID, HID);

    // 2) fused MLP (x converted inline, h1 staged in LDS)
    mlp_fused<<<(Nn + 31) / 32, 256, 0, stream>>>(x, w1t, b1, w2t, b2, h, Nn);

    // 3) AGNN
    node_rnorm<<<(Nn + 7) / 8, 256, 0, stream>>>(h, rnorm, Nn);
    zero_f32<<<1024, 256, 0, stream>>>(ssum, Nn);
    zero_f32<<<4096, 256, 0, stream>>>(hout, Nn * HID);
    int edgeBlocks = (Etot + 7) / 8;
    edge_exp<<<edgeBlocks, 256, 0, stream>>>(h, rnorm, eidx, E, Etot, beta, exbuf, ssum);
    edge_agg<<<edgeBlocks, 256, 0, stream>>>(h, eidx, E, Etot, exbuf, ssum, hout);

    // 4) classifier
    classifier<<<(Nn + 7) / 8, 256, 0, stream>>>(hout, Wc, bc, y, Nn);
}
